// MixtralSparseMoeBlock_20873541058594
// MI455X (gfx1250) — matrix operand order repaired and checked by reading, not compiled, not run on hardware
//
#include <hip/hip_runtime.h>
#include <hip/hip_bf16.h>
#include <stdint.h>

// Problem constants (from reference): B=4, S=2048 -> T=8192 tokens
#define TOK   8192
#define HDIM  1024
#define FDIM  2048
#define NEXP  8
#define TOPK  2

typedef __attribute__((ext_vector_type(16))) __bf16 bf16x16;
typedef __attribute__((ext_vector_type(8)))  __bf16 bf16x8;
typedef __attribute__((ext_vector_type(4)))  __bf16 bf16x4;
typedef __attribute__((ext_vector_type(8)))  float  f32x8;

union ABReg { bf16x16 v; bf16x8 h[2]; };

static __device__ __forceinline__ f32x8 wmma_bf16(bf16x16 a, bf16x16 b, f32x8 c) {
  // D = A(16x32 bf16) x B(32x16 bf16) + C(16x16 f32)
  return __builtin_amdgcn_wmma_f32_16x16x32_bf16(false, a, false, b, (short)0, c, false, false);
}

// 16B vector load from uniform base + 32-bit element offset
static __device__ __forceinline__ bf16x8 ldg8(const __bf16* __restrict__ base, uint32_t eoff) {
  return *(const bf16x8*)(base + eoff);
}

// ---------------------------------------------------------------------------
// fp32 -> bf16 conversion (grid-stride, float4 reads). n must be multiple of 4.
// ---------------------------------------------------------------------------
__global__ __launch_bounds__(256) void cvt_bf16_kernel(const float* __restrict__ in,
                                                       __bf16* __restrict__ out, long n4) {
  long i = (long)blockIdx.x * blockDim.x + threadIdx.x;
  long stride = (long)gridDim.x * blockDim.x;
  const float4* in4 = (const float4*)in;
  bf16x4* out4 = (bf16x4*)out;
  for (; i < n4; i += stride) {
    float4 v = in4[i];
    bf16x4 o;
    o.x = (__bf16)v.x; o.y = (__bf16)v.y; o.z = (__bf16)v.z; o.w = (__bf16)v.w;
    out4[i] = o;
  }
}

// ---------------------------------------------------------------------------
// Router: one wave (32 lanes) per token. logits = x @ gate_w^T, softmax, top-2.
// Appends (token*2+k) into per-expert gather lists via atomics.
// ---------------------------------------------------------------------------
__global__ __launch_bounds__(256) void router_kernel(const float* __restrict__ x,
                                                     const float* __restrict__ gw,
                                                     float* __restrict__ logits_out,
                                                     int* __restrict__ cnt,
                                                     int* __restrict__ entry,
                                                     float* __restrict__ wgt) {
  int wave = threadIdx.x >> 5;
  int lane = threadIdx.x & 31;
  int t = blockIdx.x * 8 + wave;
  if (t >= TOK) return;
  const float* xr = x + (size_t)t * HDIM;
  float acc[NEXP];
#pragma unroll
  for (int e = 0; e < NEXP; ++e) acc[e] = 0.f;
  for (int h = lane; h < HDIM; h += 32) {
    float xv = xr[h];
#pragma unroll
    for (int e = 0; e < NEXP; ++e) acc[e] += xv * gw[e * HDIM + h];
  }
#pragma unroll
  for (int e = 0; e < NEXP; ++e) {
#pragma unroll
    for (int off = 16; off > 0; off >>= 1) acc[e] += __shfl_xor(acc[e], off, 32);
  }
  if (lane == 0) {
    float m = acc[0];
#pragma unroll
    for (int e = 1; e < NEXP; ++e) m = fmaxf(m, acc[e]);
    float p[NEXP], s = 0.f;
#pragma unroll
    for (int e = 0; e < NEXP; ++e) { p[e] = __expf(acc[e] - m); s += p[e]; }
    float inv = 1.f / s;
#pragma unroll
    for (int e = 0; e < NEXP; ++e) p[e] *= inv;
    int i1 = 0;
#pragma unroll
    for (int e = 1; e < NEXP; ++e) if (p[e] > p[i1]) i1 = e;
    int i2 = (i1 == 0) ? 1 : 0;
#pragma unroll
    for (int e = 0; e < NEXP; ++e) if (e != i1 && p[e] > p[i2]) i2 = e;
    float den = p[i1] + p[i2];
    float w1n = p[i1] / den, w2n = p[i2] / den;
#pragma unroll
    for (int e = 0; e < NEXP; ++e) logits_out[(size_t)t * NEXP + e] = acc[e];
    int pos1 = atomicAdd(&cnt[i1], 1);
    entry[i1 * TOK + pos1] = t * 2;     wgt[i1 * TOK + pos1] = w1n;
    int pos2 = atomicAdd(&cnt[i2], 1);
    entry[i2 * TOK + pos2] = t * 2 + 1; wgt[i2 * TOK + pos2] = w2n;
  }
}

// ---------------------------------------------------------------------------
// Pass A: gated[slot, f] = silu(x@w1^T) * (x@w3^T)   (gathered per expert)
// Block: 256 thr = 8 waves; tile = [32 tokens x 256 f].
// Each wave owns [32 tokens x 32 f]: every B tile feeds 2 wmmas (mt=0,1).
// Gathered X rows double-buffered in LDS via async global->LDS loads: the
// next 256-col chunk is in flight (s_wait_asynccnt 4) while the current one
// is consumed by the wmma chain.
// ---------------------------------------------------------------------------
#define KSCHUNK 256
#define NCHUNK  (HDIM / KSCHUNK)

__global__ __launch_bounds__(256) void moe_up_kernel(const __bf16* __restrict__ xbf,
                                                     const __bf16* __restrict__ w1bf,
                                                     const __bf16* __restrict__ w3bf,
                                                     const int* __restrict__ cnt,
                                                     const int* __restrict__ entry,
                                                     __bf16* __restrict__ gated) {
  const int e = blockIdx.z;
  const int cntE = cnt[e];
  const int t0 = blockIdx.x * 32;
  if (t0 >= cntE) return;

  __shared__ __bf16 xs[2][32][KSCHUNK];   // 2 x 16 KB ping-pong
  __shared__ int sh_ent[32];

  if (threadIdx.x < 32) {
    int idx = t0 + (int)threadIdx.x;
    if (idx > cntE - 1) idx = cntE - 1;      // duplicate last row on ragged edge
    sh_ent[threadIdx.x] = entry[e * TOK + idx];
  }
  __syncthreads();

  const int wave = threadIdx.x >> 5;
  const int lane = threadIdx.x & 31;
  const int m    = lane & 15;
  const int half = lane >> 4;
  const int kA   = half * 8;          // A-matrix K offsets: {0,16}+kA
  const int kB   = half * 8;          // B-matrix K offsets: kB, kB+16
  const int fbase = blockIdx.y * 256 + wave * 32;   // 32-f slice per wave

  // uniform scalar bases + 32-bit per-lane element offsets
  const __bf16* w1e = w1bf + (size_t)e * FDIM * HDIM;
  const __bf16* w3e = w3bf + (size_t)e * FDIM * HDIM;
  uint32_t fOff[2];
#pragma unroll
  for (int nt = 0; nt < 2; ++nt)
    fOff[nt] = ((uint32_t)(fbase + nt * 16 + m) << 10) + (uint32_t)kB;  // *HDIM

  f32x8 acc1[2][2] = {}, acc3[2][2] = {};   // [mt][nt]

  // async stage of one 32x256 chunk: 1024 16B pieces / 256 thr = 4 per thread
  auto stage = [&](int buf, int ks) {
    for (int c = threadIdx.x; c < 32 * (KSCHUNK / 8); c += 256) {
      int row = c >> 5;                 // 32 pieces per row
      int col = (c & 31) * 8;           // element offset
      int tok = sh_ent[row] >> 1;
      const __bf16* src = xbf + (size_t)tok * HDIM + ks + col;
      uint32_t dst = (uint32_t)(uintptr_t)(&xs[buf][row][col]);
      asm volatile("global_load_async_to_lds_b128 %0, %1, off"
                   :: "v"(dst), "v"((uint64_t)(uintptr_t)src) : "memory");
    }
  };

  stage(0, 0);
  for (int ci = 0; ci < NCHUNK; ++ci) {
    const int cur = ci & 1;
    __syncthreads();                    // everyone done reading buffer cur^1
    if (ci + 1 < NCHUNK) {
      stage(cur ^ 1, (ci + 1) * KSCHUNK);
      asm volatile("s_wait_asynccnt 4" ::: "memory");  // older group (cur) done
    } else {
      asm volatile("s_wait_asynccnt 0" ::: "memory");
    }
    __syncthreads();

    for (int kk = 0; kk < KSCHUNK; kk += 32) {
      const uint32_t kg = (uint32_t)(ci * KSCHUNK + kk);
      ABReg a[2];
#pragma unroll
      for (int mt = 0; mt < 2; ++mt) {
        a[mt].h[0] = *(const bf16x8*)&xs[cur][mt * 16 + m][kk + kA];
        a[mt].h[1] = *(const bf16x8*)&xs[cur][mt * 16 + m][kk + kA + 16];
      }
      ABReg b1[2], b3[2];
#pragma unroll
      for (int nt = 0; nt < 2; ++nt) {
        uint32_t off = fOff[nt] + kg;
        b1[nt].h[0] = ldg8(w1e, off);
        b1[nt].h[1] = ldg8(w1e, off + 16);
        b3[nt].h[0] = ldg8(w3e, off);
        b3[nt].h[1] = ldg8(w3e, off + 16);
      }
#pragma unroll
      for (int nt = 0; nt < 2; ++nt)
#pragma unroll
        for (int mt = 0; mt < 2; ++mt) {
          acc1[mt][nt] = wmma_bf16(a[mt].v, b1[nt].v, acc1[mt][nt]);
          acc3[mt][nt] = wmma_bf16(a[mt].v, b3[nt].v, acc3[mt][nt]);
        }
    }
  }

  // epilogue: D layout -> element (mrow = v + 8*half, ncol = lane&15)
#pragma unroll
  for (int mt = 0; mt < 2; ++mt) {
#pragma unroll
    for (int nt = 0; nt < 2; ++nt) {
#pragma unroll
      for (int v = 0; v < 8; ++v) {
        int row = mt * 16 + v + 8 * half;
        if (t0 + row < cntE) {
          float a1 = acc1[mt][nt][v], a3 = acc3[mt][nt][v];
          float gv = (a1 / (1.f + __expf(-a1))) * a3;   // silu(a1) * a3
          int fcol = fbase + nt * 16 + m;
          int ent = sh_ent[row];
          gated[(size_t)ent * FDIM + fcol] = (__bf16)gv;
        }
      }
    }
  }
}

// ---------------------------------------------------------------------------
// Pass B: out[t] += w * (gated[slot] @ w2^T)   (gathered per expert, atomics)
// Block tile = [32 tokens x 256 h]; wave = [32 tokens x 32 h]; every B tile
// feeds 2 wmmas. A rows stream from the L2-resident gated buffer (bf16).
// ---------------------------------------------------------------------------
__global__ __launch_bounds__(256) void moe_down_kernel(const __bf16* __restrict__ gated,
                                                       const __bf16* __restrict__ w2bf,
                                                       const int* __restrict__ cnt,
                                                       const int* __restrict__ entry,
                                                       const float* __restrict__ wgt,
                                                       float* __restrict__ out) {
  const int e = blockIdx.z;
  const int cntE = cnt[e];
  const int t0 = blockIdx.x * 32;
  if (t0 >= cntE) return;

  __shared__ int   sh_ent[32];
  __shared__ float sh_w[32];
  if (threadIdx.x < 32) {
    int idx = t0 + (int)threadIdx.x;
    if (idx > cntE - 1) idx = cntE - 1;
    sh_ent[threadIdx.x] = entry[e * TOK + idx];
    sh_w[threadIdx.x]   = wgt[e * TOK + idx];
  }
  __syncthreads();

  const int wave = threadIdx.x >> 5;
  const int lane = threadIdx.x & 31;
  const int m    = lane & 15;
  const int half = lane >> 4;
  const int kA   = half * 8;
  const int kB   = half * 8;
  const int hbase = blockIdx.y * 256 + wave * 32;

  uint32_t arow[2];
#pragma unroll
  for (int mt = 0; mt < 2; ++mt)
    arow[mt] = (uint32_t)sh_ent[mt * 16 + m] * FDIM + (uint32_t)kA;  // <= 2^25

  const __bf16* w2e = w2bf + (size_t)e * HDIM * FDIM;
  uint32_t hOff[2];
#pragma unroll
  for (int nt = 0; nt < 2; ++nt)
    hOff[nt] = ((uint32_t)(hbase + nt * 16 + m) << 11) + (uint32_t)kB;  // *FDIM

  f32x8 acc[2][2] = {};   // [mt][nt]
  for (uint32_t kk = 0; kk < FDIM; kk += 32) {
    ABReg a[2];
#pragma unroll
    for (int mt = 0; mt < 2; ++mt) {
      a[mt].h[0] = ldg8(gated, arow[mt] + kk);
      a[mt].h[1] = ldg8(gated, arow[mt] + kk + 16);
    }
    ABReg b[2];
#pragma unroll
    for (int nt = 0; nt < 2; ++nt) {
      uint32_t off = hOff[nt] + kk;
      b[nt].h[0] = ldg8(w2e, off);
      b[nt].h[1] = ldg8(w2e, off + 16);
    }
#pragma unroll
    for (int nt = 0; nt < 2; ++nt)
#pragma unroll
      for (int mt = 0; mt < 2; ++mt)
        acc[mt][nt] = wmma_bf16(a[mt].v, b[nt].v, acc[mt][nt]);
  }

#pragma unroll
  for (int mt = 0; mt < 2; ++mt) {
#pragma unroll
    for (int nt = 0; nt < 2; ++nt) {
#pragma unroll
      for (int v = 0; v < 8; ++v) {
        int row = mt * 16 + v + 8 * half;
        if (t0 + row < cntE) {
          int tok = sh_ent[row] >> 1;
          float wv = sh_w[row];
          int h = hbase + nt * 16 + m;
          unsafeAtomicAdd(&out[(size_t)tok * HDIM + h], wv * acc[mt][nt][v]);
        }
      }
    }
  }
}

// ---------------------------------------------------------------------------
// Host launch
// ---------------------------------------------------------------------------
extern "C" void kernel_launch(void* const* d_in, const int* in_sizes, int n_in,
                              void* d_out, int out_size, void* d_ws, size_t ws_size,
                              hipStream_t stream) {
  const float* x   = (const float*)d_in[0];   // [T, H]
  const float* gw  = (const float*)d_in[1];   // [E, H]
  const float* w1  = (const float*)d_in[2];   // [E, F, H]
  const float* w3  = (const float*)d_in[3];   // [E, F, H]
  const float* w2  = (const float*)d_in[4];   // [E, H, F]
  float* out       = (float*)d_out;           // [T*H] out  + [T*E] logits

  // workspace layout (all offsets 256B aligned)
  char* ws = (char*)d_ws;
  const size_t SZ_XBF = (size_t)TOK * HDIM * 2;
  const size_t SZ_W   = (size_t)NEXP * FDIM * HDIM * 2;
  const size_t SZ_GAT = (size_t)TOK * TOPK * FDIM * 2;
  __bf16* xbf   = (__bf16*)(ws);
  __bf16* w1bf  = (__bf16*)(ws + SZ_XBF);
  __bf16* w3bf  = (__bf16*)(ws + SZ_XBF + SZ_W);
  __bf16* w2bf  = (__bf16*)(ws + SZ_XBF + 2 * SZ_W);
  __bf16* gated = (__bf16*)(ws + SZ_XBF + 3 * SZ_W);
  int*    cnt   = (int*)   (ws + SZ_XBF + 3 * SZ_W + SZ_GAT);
  int*    entry = (int*)   (ws + SZ_XBF + 3 * SZ_W + SZ_GAT + 256);
  float*  ewgt  = (float*) (ws + SZ_XBF + 3 * SZ_W + SZ_GAT + 256 + (size_t)NEXP * TOK * 4);

  // zero accumulation target + expert counters (graph-capturable memsets)
  hipMemsetAsync(out, 0, (size_t)TOK * HDIM * sizeof(float), stream);
  hipMemsetAsync(cnt, 0, NEXP * sizeof(int), stream);

  // fp32 -> bf16 conversions
  cvt_bf16_kernel<<<2048, 256, 0, stream>>>(x,  xbf,  (long)TOK * HDIM / 4);
  cvt_bf16_kernel<<<4096, 256, 0, stream>>>(w1, w1bf, (long)NEXP * FDIM * HDIM / 4);
  cvt_bf16_kernel<<<4096, 256, 0, stream>>>(w3, w3bf, (long)NEXP * FDIM * HDIM / 4);
  cvt_bf16_kernel<<<4096, 256, 0, stream>>>(w2, w2bf, (long)NEXP * HDIM * FDIM / 4);

  // router: logits + top-2 gather lists
  float* logits = out + (size_t)TOK * HDIM;
  router_kernel<<<TOK / 8, 256, 0, stream>>>(x, gw, logits, cnt, entry, ewgt);

  // pass A: up projection + silu*mul -> gated (bf16)
  moe_up_kernel<<<dim3(TOK / 32, FDIM / 256, NEXP), 256, 0, stream>>>(
      xbf, w1bf, w3bf, cnt, entry, gated);

  // pass B: down projection, scaled atomic accumulation into out
  moe_down_kernel<<<dim3(TOK / 32, HDIM / 256, NEXP), 256, 0, stream>>>(
      gated, w2bf, cnt, entry, ewgt, out);
}
